// EncoderLayer_33200097198989
// MI455X (gfx1250) — matrix operand order repaired and checked by reading, not compiled, not run on hardware
//
#include <hip/hip_runtime.h>
#include <hip/hip_bf16.h>
#include <cstdint>

// ---------------------------------------------------------------------------
// Types
// ---------------------------------------------------------------------------
typedef __attribute__((ext_vector_type(16))) __bf16 v16bf;
typedef __attribute__((ext_vector_type(8)))  __bf16 v8bf;
typedef __attribute__((ext_vector_type(8)))  float  v8f;
typedef __attribute__((ext_vector_type(4)))  __bf16 v4bf;
typedef __attribute__((ext_vector_type(4)))  unsigned int v4u;
typedef __attribute__((ext_vector_type(8)))  int v8i_;
typedef __attribute__((ext_vector_type(4)))  int v4i_;

#define B_  2
#define S_  4096
#define D_  768
#define H_  12
#define DK_ 64
#define F_  3072
#define M_  (B_ * S_)
#define EPS_ 1e-5f

// TDM availability (signature differs across toolchains; see bridge doc)
#if defined(__AMDGCN__) && __has_builtin(__builtin_amdgcn_tensor_load_to_lds) && \
    __has_builtin(__builtin_amdgcn_s_wait_tensorcnt)
#define HAS_TDM 1
#if __has_include(<hip/amd_detail/amd_gfx1250_TDM.h>)
#define TDM_6ARG 1
#endif
#endif

__device__ inline v8f wmma_bf16(const v16bf& a, const v16bf& b, const v8f& c) {
  return __builtin_amdgcn_wmma_f32_16x16x32_bf16(
      false, a, false, b, (short)0, c, false, false);
}

__device__ inline v16bf cat8(v8bf lo, v8bf hi) {
  return __builtin_shufflevector(lo, hi, 0, 1, 2, 3, 4, 5, 6, 7, 8, 9, 10, 11,
                                 12, 13, 14, 15);
}

// A-fragment (16x32, MxK) from a row-major [M][K] tile (AMD table K order: k=8h+s, then 16+8h+s-8)
__device__ inline v16bf load_a(const __bf16* __restrict__ t, int ld, int mbase,
                               int kbase) {
  const int lane = threadIdx.x & 31;
  const int hl = lane >> 4, r = lane & 15;
  const __bf16* p = t + (size_t)(mbase + r) * ld + kbase + hl * 8;
  return cat8(*(const v8bf*)p, *(const v8bf*)(p + 16));
}

// B-fragment (32x16, KxN) from a matrix stored [N][K] (AMD table K order: k=8h+s, then 16+8h+s-8)
__device__ inline v16bf load_bt(const __bf16* __restrict__ t, int ld, int kbase,
                                int nbase) {
  const int lane = threadIdx.x & 31;
  const int hl = lane >> 4, r = lane & 15;
  const __bf16* p = t + (size_t)(nbase + r) * ld + kbase + hl * 8;
  return cat8(*(const v8bf*)p, *(const v8bf*)(p + 16));
}

__device__ inline float hmax16(float x) {
#pragma unroll
  for (int m = 1; m < 16; m <<= 1) x = fmaxf(x, __shfl_xor(x, m, 32));
  return x;
}
__device__ inline float hsum16(float x) {
#pragma unroll
  for (int m = 1; m < 16; m <<= 1) x += __shfl_xor(x, m, 32);
  return x;
}

#if defined(HAS_TDM)
// Issue a TDM load of a 64-row x 32-col bf16 tile (row stride K elems) from
// global into LDS laid out with LDA_S=40 elem rows (64B data + 16B pad).
// D# packing per CDNA5 ISA sec 8.3/8.4. One wave issues; EXEC ignored.
__device__ inline void tdm_load_tile(const __bf16* gsrc, unsigned lds_addr,
                                     int K) {
  unsigned long long ga = (unsigned long long)(uintptr_t)gsrc;
  v4u g0;
  g0[0] = 1u;                                   // count=1, user mode
  g0[1] = lds_addr;                             // lds_addr [63:32]
  g0[2] = (unsigned)ga;                         // global_addr lo
  g0[3] = ((unsigned)(ga >> 32) & 0x01FFFFFFu)  // global_addr[56:32]
          | 0x80000000u;                        // type=2 ("image")
  const unsigned td0 = 1u << 20, td1 = 1u << 20;  // large tensor dims (no OOB)
  v8i_ g1;
  g1[0] = (int)((1u << 16)      // data_size = 2 bytes
                | (1u << 20)    // pad_enable
                | (3u << 22)    // pad_interval: 16 DWORDs (one 64B row)
                | (3u << 25));  // pad_amount: 4 DWORDs (16B)
  g1[1] = (int)((td0 & 0xFFFFu) << 16);
  g1[2] = (int)(((td0 >> 16) & 0xFFFFu) | ((td1 & 0xFFFFu) << 16));
  g1[3] = (int)(((td1 >> 16) & 0xFFFFu) | (32u << 16));  // tile_dim0 = 32
  g1[4] = (int)64u;                                      // tile_dim1 = 64
  g1[5] = (int)(unsigned)K;                              // dim0 stride lo32
  g1[6] = 0;                                             // stride hi16 | d1s lo
  g1[7] = 0;
  v4i_ z4 = {0, 0, 0, 0};
#if defined(TDM_6ARG)
  v8i_ z8 = {0, 0, 0, 0, 0, 0, 0, 0};
  __builtin_amdgcn_tensor_load_to_lds(g0, g1, z4, z4, z8, 0);
#else
  __builtin_amdgcn_tensor_load_to_lds(g0, g1, z4, z4, 0);
#endif
}
#endif

// ---------------------------------------------------------------------------
// fp32 -> bf16 (row-major copy)
// ---------------------------------------------------------------------------
__global__ __launch_bounds__(256) void cvt_bf16_kernel(
    const float* __restrict__ in, __bf16* __restrict__ out, int n4) {
  int i = blockIdx.x * 256 + threadIdx.x;
  if (i < n4) {
    float4 f = ((const float4*)in)[i];
    v4bf o = {(__bf16)f.x, (__bf16)f.y, (__bf16)f.z, (__bf16)f.w};
    ((v4bf*)out)[i] = o;
  }
}

// fp32 [K][N] -> bf16 [N][K] (fused convert + transpose, 32x32 LDS tiles)
__global__ __launch_bounds__(256) void cvt_t_kernel(
    const float* __restrict__ in, __bf16* __restrict__ out, int K, int N) {
  __shared__ float tile[32][33];
  const int tx = threadIdx.x & 31;
  const int ty = threadIdx.x >> 5;
  const int n0 = blockIdx.x * 32;
  const int k0 = blockIdx.y * 32;
#pragma unroll
  for (int i = 0; i < 4; ++i)
    tile[ty + i * 8][tx] = in[(size_t)(k0 + ty + i * 8) * N + n0 + tx];
  __syncthreads();
#pragma unroll
  for (int i = 0; i < 4; ++i)
    out[(size_t)(n0 + ty + i * 8) * K + k0 + tx] =
        (__bf16)tile[tx][ty + i * 8];
}

// ---------------------------------------------------------------------------
// GEMM: C[M,N] = A[M,K] @ Bt[N,K]^T + bias. Block tile 64x256, K-slice 32.
// 8 waves 2(m) x 4(n); wave tile 32x64 = 2x4 WMMA accumulators.
// A tiles streamed into LDS by the Tensor Data Mover (double-buffered,
// TENSORcnt-pipelined); B fragments read straight from L2-resident
// [N][K]-transposed weights (two b128 per lane per fragment).
// ---------------------------------------------------------------------------
#define LDA_S 40  // 32 data + 8 pad (bf16 elems) == 64B + 16B per row

template <int WF, int WB, int RELU>
__global__ __launch_bounds__(256) void gemm_bf16_kernel(
    const __bf16* __restrict__ A, const __bf16* __restrict__ Bt,
    const float* __restrict__ bias, float* __restrict__ Cf,
    __bf16* __restrict__ Cb, int M, int N, int K) {
  const int tid = threadIdx.x;
  const int wave = tid >> 5;
  const int lane = tid & 31;
  const int hl = lane >> 4, r = lane & 15;
  const int wm = (wave >> 2) * 32;
  const int wn = (wave & 3) * 64;
  const int m0 = blockIdx.x * 64;
  const int n0 = blockIdx.y * 256;

  v8f acc[2][4];
  v8f zero = {};
#pragma unroll
  for (int i = 0; i < 2; ++i)
#pragma unroll
    for (int j = 0; j < 4; ++j) acc[i][j] = zero;

#if defined(HAS_TDM)
  __shared__ __bf16 As[2][64 * LDA_S];
  const __bf16* arow = A + (size_t)m0 * K;
  if (wave == 0) tdm_load_tile(arow, (unsigned)(uintptr_t)&As[0][0], K);
  const int nk = K >> 5;
  for (int t = 0; t < nk; ++t) {
    const int k0 = t << 5;
    const int cur = t & 1;
    if (wave == 0) {
      if (t + 1 < nk) {
        tdm_load_tile(arow + k0 + 32, (unsigned)(uintptr_t)&As[cur ^ 1][0], K);
        __builtin_amdgcn_s_wait_tensorcnt(1);  // tile t landed
      } else {
        __builtin_amdgcn_s_wait_tensorcnt(0);
      }
    }
    if (t + 1 < nk)
      __builtin_prefetch(Bt + (size_t)(n0 + wn + lane) * K + k0 + 32, 0, 3);
    __syncthreads();

    v16bf af0 = load_a(As[cur], LDA_S, wm + 0, 0);
    v16bf af1 = load_a(As[cur], LDA_S, wm + 16, 0);
#pragma unroll
    for (int j = 0; j < 4; ++j) {
      v16bf bf = load_bt(Bt, K, k0, n0 + wn + j * 16);
      acc[0][j] = wmma_bf16(af0, bf, acc[0][j]);
      acc[1][j] = wmma_bf16(af1, bf, acc[1][j]);
    }
    __syncthreads();
  }
#else
  __shared__ __bf16 As[64 * LDA_S];
  const int a_row = tid >> 2;
  const int a_col = (tid & 3) * 8;
  for (int k0 = 0; k0 < K; k0 += 32) {
    *(uint4*)(As + a_row * LDA_S + a_col) =
        *(const uint4*)(A + (size_t)(m0 + a_row) * K + k0 + a_col);
    if (k0 + 32 < K)
      __builtin_prefetch(Bt + (size_t)(n0 + wn + lane) * K + k0 + 32, 0, 3);
    __syncthreads();
    v16bf af0 = load_a(As, LDA_S, wm + 0, 0);
    v16bf af1 = load_a(As, LDA_S, wm + 16, 0);
#pragma unroll
    for (int j = 0; j < 4; ++j) {
      v16bf bf = load_bt(Bt, K, k0, n0 + wn + j * 16);
      acc[0][j] = wmma_bf16(af0, bf, acc[0][j]);
      acc[1][j] = wmma_bf16(af1, bf, acc[1][j]);
    }
    __syncthreads();
  }
#endif

#pragma unroll
  for (int j = 0; j < 4; ++j) {
    const int col = n0 + wn + j * 16 + r;
    const float bv = bias[col];
#pragma unroll
    for (int i = 0; i < 2; ++i) {
#pragma unroll
      for (int v = 0; v < 8; ++v) {
        const int row = m0 + wm + i * 16 + v + 8 * hl;
        float val = acc[i][j][v] + bv;
        if (RELU) val = fmaxf(val, 0.0f);
        if (WF) Cf[(size_t)row * N + col] = val;
        if (WB) Cb[(size_t)row * N + col] = (__bf16)val;
      }
    }
  }
}

// ---------------------------------------------------------------------------
// Fused flash attention. Block = 128 threads (4 waves) = (b, h, 64 q-rows).
// ---------------------------------------------------------------------------
#define AT_LD 72  // 64 + 8 pad

__global__ __launch_bounds__(128) void attn_kernel(
    const __bf16* __restrict__ Qg, const __bf16* __restrict__ Kg,
    const __bf16* __restrict__ Vg, __bf16* __restrict__ ctx) {
  __shared__ __bf16 Qs[64 * AT_LD];     // [qrow][dk]
  __shared__ __bf16 Ks[64 * AT_LD];     // [key][dk]
  __shared__ __bf16 Vt[64 * AT_LD];     // [dk][key]
  __shared__ __bf16 Ps[4][16 * AT_LD];  // per-wave P [qrow][key]

  const int tid = threadIdx.x;
  const int wave = tid >> 5;
  const int lane = tid & 31;
  const int hl = lane >> 4, r = lane & 15;
  const int q0 = blockIdx.x * 64;
  const int h = blockIdx.y;
  const int b = blockIdx.z;

  const int c_row = tid >> 1;
  const int c_col = (tid & 1) * 32;
#pragma unroll
  for (int u = 0; u < 4; ++u) {
    *(uint4*)(Qs + c_row * AT_LD + c_col + u * 8) =
        *(const uint4*)(Qg + ((size_t)b * S_ + q0 + c_row) * D_ + h * DK_ +
                        c_col + u * 8);
  }

  float m_i[8], l_i[8];
  v8f o_acc[4];
  v8f zero = {};
#pragma unroll
  for (int v = 0; v < 8; ++v) { m_i[v] = -3.0e38f; l_i[v] = 0.0f; }
#pragma unroll
  for (int jd = 0; jd < 4; ++jd) o_acc[jd] = zero;

  __syncthreads();

  for (int kb = 0; kb < S_ / 64; ++kb) {
    const size_t gbase = ((size_t)b * S_ + kb * 64 + c_row) * D_ + h * DK_;
#pragma unroll
    for (int u = 0; u < 4; ++u) {
      *(uint4*)(Ks + c_row * AT_LD + c_col + u * 8) =
          *(const uint4*)(Kg + gbase + c_col + u * 8);
      v8bf vv = *(const v8bf*)(Vg + gbase + c_col + u * 8);
#pragma unroll
      for (int e = 0; e < 8; ++e)
        Vt[(c_col + u * 8 + e) * AT_LD + c_row] = vv[e];
    }
    __syncthreads();

    v16bf aq0 = load_a(Qs, AT_LD, wave * 16, 0);
    v16bf aq1 = load_a(Qs, AT_LD, wave * 16, 32);
    v8f s_acc[4];
#pragma unroll
    for (int j = 0; j < 4; ++j) {
      s_acc[j] = zero;
      s_acc[j] = wmma_bf16(aq0, load_bt(Ks, AT_LD, 0, j * 16), s_acc[j]);
      s_acc[j] = wmma_bf16(aq1, load_bt(Ks, AT_LD, 32, j * 16), s_acc[j]);
    }
#pragma unroll
    for (int j = 0; j < 4; ++j)
#pragma unroll
      for (int v = 0; v < 8; ++v) s_acc[j][v] *= 0.125f;  // 1/sqrt(64)

#pragma unroll
    for (int v = 0; v < 8; ++v) {
      float mloc = fmaxf(fmaxf(s_acc[0][v], s_acc[1][v]),
                         fmaxf(s_acc[2][v], s_acc[3][v]));
      float mnew = fmaxf(m_i[v], hmax16(mloc));
      float scale = __expf(m_i[v] - mnew);
      float rsum = 0.0f;
#pragma unroll
      for (int j = 0; j < 4; ++j) {
        float p = __expf(s_acc[j][v] - mnew);
        rsum += p;
        Ps[wave][(v + 8 * hl) * AT_LD + j * 16 + r] = (__bf16)p;
      }
      l_i[v] = l_i[v] * scale + hsum16(rsum);
      m_i[v] = mnew;
#pragma unroll
      for (int jd = 0; jd < 4; ++jd) o_acc[jd][v] *= scale;
    }

    v16bf ap0 = load_a(Ps[wave], AT_LD, 0, 0);
    v16bf ap1 = load_a(Ps[wave], AT_LD, 0, 32);
#pragma unroll
    for (int jd = 0; jd < 4; ++jd) {
      o_acc[jd] = wmma_bf16(ap0, load_bt(Vt, AT_LD, 0, jd * 16), o_acc[jd]);
      o_acc[jd] = wmma_bf16(ap1, load_bt(Vt, AT_LD, 32, jd * 16), o_acc[jd]);
    }
    __syncthreads();
  }

#pragma unroll
  for (int jd = 0; jd < 4; ++jd)
#pragma unroll
    for (int v = 0; v < 8; ++v) {
      const int row = q0 + wave * 16 + v + 8 * hl;
      const int col = h * DK_ + jd * 16 + r;
      ctx[((size_t)b * S_ + row) * D_ + col] = (__bf16)(o_acc[jd][v] / l_i[v]);
    }
}

// ---------------------------------------------------------------------------
// Residual add + LayerNorm (unbiased std, (x-m)/(std+eps)); bf16 mirror opt.
// ---------------------------------------------------------------------------
template <int WB>
__global__ __launch_bounds__(256) void add_ln_kernel(
    const float* __restrict__ X, const float* __restrict__ Y,
    const float* __restrict__ alpha, const float* __restrict__ gamma,
    float* __restrict__ Of, __bf16* __restrict__ Ob) {
  __shared__ float red[2][8];
  const int row = blockIdx.x;
  const int tid = threadIdx.x;
  const int wv = tid >> 5, lane = tid & 31;
  const float* xr = X + (size_t)row * D_;
  const float* yr = Y + (size_t)row * D_;

  float vals[3];
  float s = 0.0f, ss = 0.0f;
#pragma unroll
  for (int i = 0; i < 3; ++i) {
    const int c = tid + i * 256;
    float v = xr[c] + yr[c];
    vals[i] = v;
    s += v;
    ss += v * v;
  }
#pragma unroll
  for (int m = 16; m >= 1; m >>= 1) {
    s += __shfl_xor(s, m, 32);
    ss += __shfl_xor(ss, m, 32);
  }
  if (lane == 0) { red[0][wv] = s; red[1][wv] = ss; }
  __syncthreads();
  if (tid == 0) {
    float ts = 0.0f, tss = 0.0f;
#pragma unroll
    for (int w = 0; w < 8; ++w) { ts += red[0][w]; tss += red[1][w]; }
    red[0][0] = ts;
    red[1][0] = tss;
  }
  __syncthreads();
  const float mean = red[0][0] * (1.0f / D_);
  float var = (red[1][0] - (float)D_ * mean * mean) * (1.0f / (D_ - 1));
  var = fmaxf(var, 0.0f);
  const float k = alpha[0] / (sqrtf(var) + EPS_);
  const float g = gamma[0];
#pragma unroll
  for (int i = 0; i < 3; ++i) {
    const int c = tid + i * 256;
    float o = (vals[i] - mean) * k + g;
    Of[(size_t)row * D_ + c] = o;
    if (WB) Ob[(size_t)row * D_ + c] = (__bf16)o;
  }
}

// ---------------------------------------------------------------------------
// Host orchestration
// ---------------------------------------------------------------------------
extern "C" void kernel_launch(void* const* d_in, const int* in_sizes, int n_in,
                              void* d_out, int out_size, void* d_ws,
                              size_t ws_size, hipStream_t stream) {
  (void)in_sizes; (void)n_in; (void)out_size; (void)ws_size;

  const float* x  = (const float*)d_in[0];
  const float* wq = (const float*)d_in[1];
  const float* bq = (const float*)d_in[2];
  const float* wk = (const float*)d_in[3];
  const float* bk = (const float*)d_in[4];
  const float* wv = (const float*)d_in[5];
  const float* bv = (const float*)d_in[6];
  const float* wo = (const float*)d_in[7];
  const float* bo = (const float*)d_in[8];
  const float* w1 = (const float*)d_in[9];
  const float* b1 = (const float*)d_in[10];
  const float* w2 = (const float*)d_in[11];
  const float* b2 = (const float*)d_in[12];
  const float* a1 = (const float*)d_in[13];
  const float* g1 = (const float*)d_in[14];
  const float* a2 = (const float*)d_in[15];
  const float* g2 = (const float*)d_in[16];
  float* out = (float*)d_out;

  char* ws = (char*)d_ws;
  size_t off = 0;
  auto carve = [&](size_t bytes) -> char* {
    char* p = ws + off;
    off = (off + bytes + 255) & ~(size_t)255;
    return p;
  };

  __bf16* x_bf   = (__bf16*)carve((size_t)M_ * D_ * 2);
  __bf16* wq_t   = (__bf16*)carve((size_t)D_ * D_ * 2);  // [N][K]
  __bf16* wk_t   = (__bf16*)carve((size_t)D_ * D_ * 2);
  __bf16* wv_t   = (__bf16*)carve((size_t)D_ * D_ * 2);
  __bf16* wo_t   = (__bf16*)carve((size_t)D_ * D_ * 2);
  __bf16* w1_t   = (__bf16*)carve((size_t)D_ * F_ * 2);  // [F][D]
  __bf16* w2_t   = (__bf16*)carve((size_t)F_ * D_ * 2);  // [D][F]
  __bf16* q_bf   = (__bf16*)carve((size_t)M_ * D_ * 2);
  __bf16* k_bf   = (__bf16*)carve((size_t)M_ * D_ * 2);
  __bf16* v_bf   = (__bf16*)carve((size_t)M_ * D_ * 2);
  __bf16* ctx_bf = (__bf16*)carve((size_t)M_ * D_ * 2);
  __bf16* h1_bf  = (__bf16*)carve((size_t)M_ * D_ * 2);
  __bf16* ff1_bf = (__bf16*)carve((size_t)M_ * F_ * 2);
  float*  attn_f = (float*)carve((size_t)M_ * D_ * 4);
  float*  h1_f   = (float*)carve((size_t)M_ * D_ * 4);
  float*  ff2_f  = (float*)carve((size_t)M_ * D_ * 4);

  {
    int n4 = (int)((size_t)M_ * D_ / 4);
    cvt_bf16_kernel<<<(n4 + 255) / 256, 256, 0, stream>>>(x, x_bf, n4);
  }
  auto cvt_t = [&](const float* src, __bf16* dst, int K, int N) {
    cvt_t_kernel<<<dim3(N / 32, K / 32), 256, 0, stream>>>(src, dst, K, N);
  };
  cvt_t(wq, wq_t, D_, D_);
  cvt_t(wk, wk_t, D_, D_);
  cvt_t(wv, wv_t, D_, D_);
  cvt_t(wo, wo_t, D_, D_);
  cvt_t(w1, w1_t, D_, F_);
  cvt_t(w2, w2_t, F_, D_);

  dim3 gD(M_ / 64, D_ / 256);
  gemm_bf16_kernel<0, 1, 0><<<gD, 256, 0, stream>>>(x_bf, wq_t, bq, nullptr,
                                                    q_bf, M_, D_, D_);
  gemm_bf16_kernel<0, 1, 0><<<gD, 256, 0, stream>>>(x_bf, wk_t, bk, nullptr,
                                                    k_bf, M_, D_, D_);
  gemm_bf16_kernel<0, 1, 0><<<gD, 256, 0, stream>>>(x_bf, wv_t, bv, nullptr,
                                                    v_bf, M_, D_, D_);

  attn_kernel<<<dim3(S_ / 64, H_, B_), 128, 0, stream>>>(q_bf, k_bf, v_bf,
                                                         ctx_bf);

  gemm_bf16_kernel<1, 0, 0><<<gD, 256, 0, stream>>>(ctx_bf, wo_t, bo, attn_f,
                                                    nullptr, M_, D_, D_);

  add_ln_kernel<1><<<M_, 256, 0, stream>>>(x, attn_f, a1, g1, h1_f, h1_bf);

  dim3 gF(M_ / 64, F_ / 256);
  gemm_bf16_kernel<0, 1, 1><<<gF, 256, 0, stream>>>(h1_bf, w1_t, b1, nullptr,
                                                    ff1_bf, M_, F_, D_);
  gemm_bf16_kernel<1, 0, 0><<<gD, 256, 0, stream>>>(ff1_bf, w2_t, b2, ff2_f,
                                                    nullptr, M_, D_, F_);

  add_ln_kernel<0><<<M_, 256, 0, stream>>>(h1_f, ff2_f, a2, g2, out, nullptr);
}
